// ComplexSSMState_4037269258854
// MI455X (gfx1250) — matrix operand order repaired and checked by reading, not compiled, not run on hardware
//
#include <hip/hip_runtime.h>

// Problem constants (from reference)
#define Bb   8
#define Ss   4096
#define Gg   4
#define Dd   128
#define Nn   128
#define TT   64                 // chunk length (tokens)
#define NCH  (Ss / TT)          // 64 chunks per (b,g)
#define NBG  (Bb * Gg)          // 32
#define BSGD (Bb * Ss * Gg * Dd)

typedef __attribute__((ext_vector_type(16))) __bf16       v16bf;
typedef __attribute__((ext_vector_type(8)))  float        v8f;
typedef __attribute__((ext_vector_type(8)))  unsigned int v8u;
typedef __attribute__((ext_vector_type(4)))  unsigned int v4u;

// Global-address-space vector type: force global_load_b128 (not flat_load)
using GV8 = const __attribute__((address_space(1))) v8u;

// f32 -> bf16 round-to-nearest-even (bit trick), and back
__device__ __forceinline__ unsigned short f2bf(float f) {
  unsigned int x = __float_as_uint(f);
  x += 0x7fffu + ((x >> 16) & 1u);
  return (unsigned short)(x >> 16);
}
__device__ __forceinline__ float bf2f(unsigned short h) {
  return __uint_as_float(((unsigned int)h) << 16);
}

// ---------------------------------------------------------------------------
// LDS layout for phases 1 and 3 (~200.5 KB, fits 320 KB WGP budget)
// Xhi/Xlo pitch 264 shorts = 132 dwords  -> conflict-free A-operand gather
// BX pitch 260 floats                    -> conflict-free WMMA D-stores
// ---------------------------------------------------------------------------
struct SM {
  unsigned short Xhi[TT][264];   // bf16 high parts of [x_r | x_i] rows (later: h)
  unsigned short Xlo[TT][264];   // bf16 low (residual) parts
  float BX[TT][260];             // GEMM output [Bx_r | Bx_i]; scan overwrites with h
  float Ar[TT][128];
  float Ai[TT][128];
  float dtw0[256];
  float dtw1[256];
  float nlA[128];
  float Aph[128];
  float dt_mag[TT];
  float dt_phase[TT];
  float dtpart[TT][4][2];
};

// A-operand fetch: two aligned 16B LDS vectors -> ds_load_b128 x2
struct V4x2 { v4u a, b; };
__device__ __forceinline__ v16bf ld_a16(const unsigned* row, int off) {
  const v4u* p = (const v4u*)(row + off);   // off is a multiple of 4 dwords
  V4x2 t{p[0], p[2]};                       // dwords [off..off+3], [off+8..off+11]
  return __builtin_bit_cast(v16bf, t);
}

#define WMMA_BF16(A, Bm, C) \
  __builtin_amdgcn_wmma_f32_16x16x32_bf16(false, (A), false, (Bm), (short)0, (C), false, false)

// ---------------------------------------------------------------------------
// 4 M-tiles x 2 N-tiles of a 64x256x256 split-bf16 GEMM (one wave).
// Weights loaded once per k-step (global_load_b128 clauses) and reused across
// 4 M-tiles; each A hi/lo pair (2x ds_load_b128) reused across both N-tiles.
// -> 2 ds_load_b128 per 3 WMMAs; 8 independent accumulator chains interleave.
// ---------------------------------------------------------------------------
__device__ __forceinline__ void gemm_oct(const unsigned* Xh32, const unsigned* Xl32,
                                         const unsigned* __restrict__ Whi,
                                         const unsigned* __restrict__ Wlo,
                                         int nt0, int lane, v8f acc[8]) {
  const int h  = lane >> 4;
  const int ml = lane & 15;
  v8f z = {};
#pragma unroll
  for (int i = 0; i < 8; ++i) acc[i] = z;
#pragma unroll
  for (int kt = 0; kt < 8; ++kt) {
    GV8* bh0 = (GV8*)(unsigned long long)(Whi + (kt * 16 + nt0) * 256);
    GV8* bl0 = (GV8*)(unsigned long long)(Wlo + (kt * 16 + nt0) * 256);
    GV8* bh1 = (GV8*)(unsigned long long)(Whi + (kt * 16 + nt0 + 1) * 256);
    GV8* bl1 = (GV8*)(unsigned long long)(Wlo + (kt * 16 + nt0 + 1) * 256);
    v16bf WH0 = __builtin_bit_cast(v16bf, bh0[lane]);
    v16bf WL0 = __builtin_bit_cast(v16bf, bl0[lane]);
    v16bf WH1 = __builtin_bit_cast(v16bf, bh1[lane]);
    v16bf WL1 = __builtin_bit_cast(v16bf, bl1[lane]);
#pragma unroll
    for (int mt = 0; mt < 4; ++mt) {
      const int ro = (mt * 16 + ml) * 132 + kt * 16 + 4 * h;
      v16bf AH = ld_a16(Xh32, ro);
      v16bf AL = ld_a16(Xl32, ro);
      acc[2 * mt]     = WMMA_BF16(AH, WH0, acc[2 * mt]);
      acc[2 * mt + 1] = WMMA_BF16(AH, WH1, acc[2 * mt + 1]);
      acc[2 * mt]     = WMMA_BF16(AH, WL0, acc[2 * mt]);
      acc[2 * mt + 1] = WMMA_BF16(AH, WL1, acc[2 * mt + 1]);
      acc[2 * mt]     = WMMA_BF16(AL, WH0, acc[2 * mt]);
      acc[2 * mt + 1] = WMMA_BF16(AL, WH1, acc[2 * mt + 1]);
    }
  }
}

// ---------------------------------------------------------------------------
// Weight element generators for the folded complex GEMMs.
// W_B (K=2D rows, N'=2N cols):  [Bx_r | Bx_i] = [x_r | x_i] * W_B
// W_C (K=2N rows, N'=2D cols):  [y_r  | y_i ] = [h_r | h_i] * W_C
// ---------------------------------------------------------------------------
__device__ __forceinline__ float wB(int k, int c, const float* Bw_r, const float* Bw_i) {
  int d = k & 127, n = c & 127;
  if (c < 128) return (k < 128) ? Bw_r[n * 128 + d] : -Bw_i[n * 128 + d];
  else         return (k < 128) ? Bw_i[n * 128 + d] :  Bw_r[n * 128 + d];
}
__device__ __forceinline__ float wC(int k, int c, const float* Cw_r, const float* Cw_i) {
  int n = k & 127, d = c & 127;
  if (c < 128) return (k < 128) ? Cw_r[d * 128 + n] : -Cw_i[d * 128 + n];
  else         return (k < 128) ? Cw_i[d * 128 + n] :  Cw_r[d * 128 + n];
}

// ---------------------------------------------------------------------------
// Phase 0: swizzle weights into hi/lo bf16 WMMA-B-operand layout; -softplus(A)
// dword index inside a tile: lane*8 + j ; element K = 8*(lane/16) + 2j (+8 if j>=4) (+e)
// ---------------------------------------------------------------------------
__global__ void __launch_bounds__(256) prep_kernel(
    const float* __restrict__ logAmag,
    const float* __restrict__ Bw_r, const float* __restrict__ Bw_i,
    const float* __restrict__ Cw_r, const float* __restrict__ Cw_i,
    unsigned* __restrict__ WBhi, unsigned* __restrict__ WBlo,
    unsigned* __restrict__ WChi, unsigned* __restrict__ WClo,
    float* __restrict__ nlA) {
  int tid = blockIdx.x * 256 + threadIdx.x;
  if (tid < Gg * Nn) {  // neg_log_A = -softplus(log_A_mag)
    float v  = logAmag[tid];
    float sp = (v > 20.0f) ? v : log1pf(__expf(v));
    nlA[tid] = -sp;
  }
  if (tid >= 2 * 32768) return;
  int mat  = tid >> 15;          // 0 = W_B, 1 = W_C
  int off  = tid & 32767;        // dword index inside the swizzled matrix
  int tile = off >> 8;           // (kt*16 + nt), 128 tiles of 256 dwords
  int lane = (off >> 3) & 31;
  int j    = off & 7;
  int kt = tile >> 4, nt = tile & 15;
  int h = lane >> 4, nl = lane & 15;
  int c  = nt * 16 + nl;
  int k0 = kt * 32 + 8 * h + 2 * j + ((j >= 4) ? 8 : 0);
  float f0, f1;
  if (mat == 0) { f0 = wB(k0, c, Bw_r, Bw_i); f1 = wB(k0 + 1, c, Bw_r, Bw_i); }
  else          { f0 = wC(k0, c, Cw_r, Cw_i); f1 = wC(k0 + 1, c, Cw_r, Cw_i); }
  unsigned short h0 = f2bf(f0), h1 = f2bf(f1);
  unsigned short l0 = f2bf(f0 - bf2f(h0)), l1 = f2bf(f1 - bf2f(h1));
  unsigned hid = ((unsigned)h1 << 16) | (unsigned)h0;
  unsigned lod = ((unsigned)l1 << 16) | (unsigned)l0;
  if (mat == 0) { WBhi[off] = hid; WBlo[off] = lod; }
  else          { WChi[off] = hid; WClo[off] = lod; }
}

// ---------------------------------------------------------------------------
// Shared front half of phases 1 and 3:
// stage x -> LDS (hi/lo bf16), compute dt, A matrices, B-projection GEMM.
// Leaves [Bx_r|Bx_i] (unscaled) in sm.BX; caller must __syncthreads() after.
// ---------------------------------------------------------------------------
__device__ __forceinline__ void front_half(
    SM& sm,
    const float* __restrict__ x_r, const float* __restrict__ x_i,
    const float* __restrict__ Aphase,
    const float* __restrict__ dt_w, const float* __restrict__ dt_b,
    const unsigned* __restrict__ WBhi, const unsigned* __restrict__ WBlo,
    const float* __restrict__ nlA,
    int g, int tid, size_t base) {
  // ---- stage chunk of x as bf16 hi/lo: float4 loads, packed b64 LDS stores ----
  for (int idx = tid; idx < TT * 64; idx += 256) {
    int t  = idx >> 6;
    int q4 = (idx & 63) << 2;
    const float* src = (q4 < 128) ? (x_r + base + (size_t)t * (Gg * Dd) + q4)
                                  : (x_i + base + (size_t)t * (Gg * Dd) + (q4 - 128));
    float4 v = *(const float4*)src;
    unsigned short h0 = f2bf(v.x), h1 = f2bf(v.y), h2 = f2bf(v.z), h3 = f2bf(v.w);
    unsigned short l0 = f2bf(v.x - bf2f(h0)), l1 = f2bf(v.y - bf2f(h1));
    unsigned short l2 = f2bf(v.z - bf2f(h2)), l3 = f2bf(v.w - bf2f(h3));
    uint2 ph = make_uint2(((unsigned)h1 << 16) | h0, ((unsigned)h3 << 16) | h2);
    uint2 pl = make_uint2(((unsigned)l1 << 16) | l0, ((unsigned)l3 << 16) | l2);
    *(uint2*)&sm.Xhi[t][q4] = ph;
    *(uint2*)&sm.Xlo[t][q4] = pl;
  }
  if (tid < 256) { sm.dtw0[tid] = dt_w[tid]; sm.dtw1[tid] = dt_w[256 + tid]; }
  if (tid < 128) { sm.nlA[tid] = nlA[g * 128 + tid]; sm.Aph[tid] = Aphase[g * 128 + tid]; }
  __syncthreads();

  // ---- dt partial dot products: 4 threads per token, vectorized LDS reads ----
  {
    int t = tid >> 2, q = tid & 3;
    float s0 = 0.0f, s1 = 0.0f;
    int k0 = q * 64;
#pragma unroll 4
    for (int kq = 0; kq < 16; ++kq) {
      int kk = k0 + kq * 4;
      uint2 ph = *(const uint2*)&sm.Xhi[t][kk];
      uint2 pl = *(const uint2*)&sm.Xlo[t][kk];
      float x0 = __uint_as_float(ph.x << 16)         + __uint_as_float(pl.x << 16);
      float x1 = __uint_as_float(ph.x & 0xffff0000u) + __uint_as_float(pl.x & 0xffff0000u);
      float x2 = __uint_as_float(ph.y << 16)         + __uint_as_float(pl.y << 16);
      float x3 = __uint_as_float(ph.y & 0xffff0000u) + __uint_as_float(pl.y & 0xffff0000u);
      s0 += x0 * sm.dtw0[kk] + x1 * sm.dtw0[kk + 1] + x2 * sm.dtw0[kk + 2] + x3 * sm.dtw0[kk + 3];
      s1 += x0 * sm.dtw1[kk] + x1 * sm.dtw1[kk + 1] + x2 * sm.dtw1[kk + 2] + x3 * sm.dtw1[kk + 3];
    }
    sm.dtpart[t][q][0] = s0;
    sm.dtpart[t][q][1] = s1;
  }
  __syncthreads();
  if (tid < TT) {
    float z0 = sm.dtpart[tid][0][0] + sm.dtpart[tid][1][0] + sm.dtpart[tid][2][0] +
               sm.dtpart[tid][3][0] + dt_b[0];
    float z1 = sm.dtpart[tid][0][1] + sm.dtpart[tid][1][1] + sm.dtpart[tid][2][1] +
               sm.dtpart[tid][3][1] + dt_b[1];
    sm.dt_mag[tid]   = fminf(fmaxf(__expf(z0), 1e-4f), 2.0f);
    sm.dt_phase[tid] = fminf(fmaxf(__expf(z1), 1e-4f), 2.0f);
  }
  __syncthreads();

  // ---- A_r, A_i via exp / sincos (trans units) ----
  for (int idx = tid; idx < TT * 128; idx += 256) {
    int t = idx >> 7, n = idx & 127;
    float am  = __expf(sm.dt_mag[t] * sm.nlA[n]);
    float ang = sm.dt_phase[t] * sm.Aph[n];
    float sa, ca;
    __sincosf(ang, &sa, &ca);
    sm.Ar[t][n] = am * ca;
    sm.Ai[t][n] = am * sa;
  }

  // ---- B-projection GEMM: 64x256x256 split-bf16; wave -> 2 N-tiles x 4 M-tiles ----
  {
    int wave = tid >> 5, lane = tid & 31;
    int h = lane >> 4, ml = lane & 15;
    int nt0 = wave * 2;
    v8f acc[8];
    gemm_oct((const unsigned*)&sm.Xhi[0][0], (const unsigned*)&sm.Xlo[0][0],
             WBhi, WBlo, nt0, lane, acc);
#pragma unroll
    for (int mt = 0; mt < 4; ++mt)
#pragma unroll
      for (int v = 0; v < 8; ++v) {
        sm.BX[mt * 16 + v + 8 * h][nt0 * 16 + ml]       = acc[2 * mt][v];
        sm.BX[mt * 16 + v + 8 * h][(nt0 + 1) * 16 + ml] = acc[2 * mt + 1][v];
      }
  }
}

// ---------------------------------------------------------------------------
// Phase 1: per-chunk local scan -> (A-product, local final h) summaries
// ---------------------------------------------------------------------------
__global__ void __launch_bounds__(256) phase1_kernel(
    const float* __restrict__ x_r, const float* __restrict__ x_i,
    const float* __restrict__ Aphase,
    const float* __restrict__ dt_w, const float* __restrict__ dt_b,
    const unsigned* __restrict__ WBhi, const unsigned* __restrict__ WBlo,
    const float* __restrict__ nlA,
    float* __restrict__ chunkP, float* __restrict__ chunkH) {
  extern __shared__ char smem_raw[];
  SM& sm = *reinterpret_cast<SM*>(smem_raw);
  int bg = blockIdx.x / NCH, chunk = blockIdx.x % NCH;
  int b = bg >> 2, g = bg & 3;
  int tid = threadIdx.x;
  size_t base = (((size_t)b * Ss + (size_t)chunk * TT) * Gg + g) * Dd;

  front_half(sm, x_r, x_i, Aphase, dt_w, dt_b, WBhi, WBlo, nlA, g, tid, base);
  __syncthreads();

  if (tid < 128) {
    int n = tid;
    float hr = 0.0f, hj = 0.0f, pr = 1.0f, pi = 0.0f;
#pragma unroll 4
    for (int t = 0; t < TT; ++t) {
      float ar = sm.Ar[t][n], ai = sm.Ai[t][n];
      float dm  = sm.dt_mag[t];
      float bxr = sm.BX[t][n] * dm;
      float bxi = sm.BX[t][128 + n] * dm;
      float nhr = ar * hr - ai * hj + bxr;
      float nhi = ar * hj + ai * hr + bxi;
      hr = nhr; hj = nhi;
      float npr = ar * pr - ai * pi;
      float npi = ar * pi + ai * pr;
      pr = npr; pi = npi;
    }
    size_t o = (size_t)blockIdx.x * 256 + n;
    chunkP[o] = pr; chunkP[o + 128] = pi;
    chunkH[o] = hr; chunkH[o + 128] = hj;
  }
}

// ---------------------------------------------------------------------------
// Phase 2: exclusive scan across chunk summaries -> per-chunk initial state
// ---------------------------------------------------------------------------
__global__ void __launch_bounds__(128) phase2_kernel(
    const float* __restrict__ chunkP, const float* __restrict__ chunkH,
    float* __restrict__ hInit) {
  int bg = blockIdx.x;
  int n  = threadIdx.x;  // 128
  float hr = 0.0f, hj = 0.0f;
  for (int c = 0; c < NCH; ++c) {
    size_t o = ((size_t)bg * NCH + c) * 256 + n;
    hInit[o] = hr; hInit[o + 128] = hj;
    float pr = chunkP[o], pi = chunkP[o + 128];
    float fr = chunkH[o], fi = chunkH[o + 128];
    float nhr = pr * hr - pi * hj + fr;
    float nhi = pr * hj + pi * hr + fi;
    hr = nhr; hj = nhi;
  }
}

// ---------------------------------------------------------------------------
// Phase 3: recompute Bx/A, scan with true initial state, C-projection -> y
// ---------------------------------------------------------------------------
__global__ void __launch_bounds__(256) phase3_kernel(
    const float* __restrict__ x_r, const float* __restrict__ x_i,
    const float* __restrict__ Aphase,
    const float* __restrict__ dt_w, const float* __restrict__ dt_b,
    const unsigned* __restrict__ WBhi, const unsigned* __restrict__ WBlo,
    const unsigned* __restrict__ WChi, const unsigned* __restrict__ WClo,
    const float* __restrict__ nlA, const float* __restrict__ hInit,
    float* __restrict__ yout) {
  extern __shared__ char smem_raw[];
  SM& sm = *reinterpret_cast<SM*>(smem_raw);
  int bg = blockIdx.x / NCH, chunk = blockIdx.x % NCH;
  int b = bg >> 2, g = bg & 3;
  int tid = threadIdx.x;
  size_t base = (((size_t)b * Ss + (size_t)chunk * TT) * Gg + g) * Dd;

  front_half(sm, x_r, x_i, Aphase, dt_w, dt_b, WBhi, WBlo, nlA, g, tid, base);
  __syncthreads();

  // scan with initial state from phase 2; h overwrites Bx in place
  if (tid < 128) {
    int n = tid;
    size_t o = (size_t)blockIdx.x * 256 + n;
    float hr = hInit[o], hj = hInit[o + 128];
#pragma unroll 4
    for (int t = 0; t < TT; ++t) {
      float ar = sm.Ar[t][n], ai = sm.Ai[t][n];
      float dm  = sm.dt_mag[t];
      float bxr = sm.BX[t][n] * dm;
      float bxi = sm.BX[t][128 + n] * dm;
      float nhr = ar * hr - ai * hj + bxr;
      float nhi = ar * hj + ai * hr + bxi;
      hr = nhr; hj = nhi;
      sm.BX[t][n]       = nhr;
      sm.BX[t][128 + n] = nhi;
    }
  }
  __syncthreads();

  // convert h (f32 in BX) -> bf16 hi/lo, reusing the X staging buffers
  for (int idx = tid; idx < TT * 64; idx += 256) {
    int t  = idx >> 6;
    int q4 = (idx & 63) << 2;
    float4 v = *(const float4*)&sm.BX[t][q4];
    unsigned short h0 = f2bf(v.x), h1 = f2bf(v.y), h2 = f2bf(v.z), h3 = f2bf(v.w);
    unsigned short l0 = f2bf(v.x - bf2f(h0)), l1 = f2bf(v.y - bf2f(h1));
    unsigned short l2 = f2bf(v.z - bf2f(h2)), l3 = f2bf(v.w - bf2f(h3));
    uint2 ph = make_uint2(((unsigned)h1 << 16) | h0, ((unsigned)h3 << 16) | h2);
    uint2 pl = make_uint2(((unsigned)l1 << 16) | l0, ((unsigned)l3 << 16) | l2);
    *(uint2*)&sm.Xhi[t][q4] = ph;
    *(uint2*)&sm.Xlo[t][q4] = pl;
  }
  __syncthreads();

  // C-projection GEMM; write y directly from WMMA accumulators
  {
    int wave = tid >> 5, lane = tid & 31;
    int h = lane >> 4, ml = lane & 15;
    int nt0 = wave * 2;
    v8f acc[8];
    gemm_oct((const unsigned*)&sm.Xhi[0][0], (const unsigned*)&sm.Xlo[0][0],
             WChi, WClo, nt0, lane, acc);
#pragma unroll
    for (int ntidx = 0; ntidx < 2; ++ntidx) {
      int c = (nt0 + ntidx) * 16 + ml;
      float* ob = (c < 128) ? (yout + c) : (yout + (size_t)BSGD + (c - 128));
#pragma unroll
      for (int mt = 0; mt < 4; ++mt)
#pragma unroll
        for (int v = 0; v < 8; ++v)
          ob[base + (size_t)(mt * 16 + v + 8 * h) * (Gg * Dd)] = acc[2 * mt + ntidx][v];
    }
  }
}

// ---------------------------------------------------------------------------
// Host launcher
// ---------------------------------------------------------------------------
extern "C" void kernel_launch(void* const* d_in, const int* in_sizes, int n_in,
                              void* d_out, int out_size, void* d_ws, size_t ws_size,
                              hipStream_t stream) {
  const float* x_r     = (const float*)d_in[0];
  const float* x_i     = (const float*)d_in[1];
  const float* logAmag = (const float*)d_in[2];
  const float* Aphase  = (const float*)d_in[3];
  const float* Bw_r    = (const float*)d_in[4];
  const float* Bw_i    = (const float*)d_in[5];
  const float* Cw_r    = (const float*)d_in[6];
  const float* Cw_i    = (const float*)d_in[7];
  const float* dt_w    = (const float*)d_in[8];
  const float* dt_b    = (const float*)d_in[9];
  float* yout = (float*)d_out;

  // Workspace layout (bytes)
  char* ws = (char*)d_ws;
  unsigned* WBhi = (unsigned*)(ws + 0);
  unsigned* WBlo = (unsigned*)(ws + 131072);
  unsigned* WChi = (unsigned*)(ws + 262144);
  unsigned* WClo = (unsigned*)(ws + 393216);
  float* nlA     = (float*)(ws + 524288);
  float* chunkP  = (float*)(ws + 526336);             // 2 MB
  float* chunkH  = (float*)(ws + 526336 + 2097152);   // 2 MB
  float* hInit   = (float*)(ws + 526336 + 4194304);   // 2 MB

  int smBytes = (int)sizeof(SM);
  (void)hipFuncSetAttribute((const void*)phase1_kernel,
                            hipFuncAttributeMaxDynamicSharedMemorySize, smBytes);
  (void)hipFuncSetAttribute((const void*)phase3_kernel,
                            hipFuncAttributeMaxDynamicSharedMemorySize, smBytes);

  prep_kernel<<<256, 256, 0, stream>>>(logAmag, Bw_r, Bw_i, Cw_r, Cw_i,
                                       WBhi, WBlo, WChi, WClo, nlA);
  phase1_kernel<<<NBG * NCH, 256, smBytes, stream>>>(x_r, x_i, Aphase, dt_w, dt_b,
                                                     WBhi, WBlo, nlA, chunkP, chunkH);
  phase2_kernel<<<NBG, 128, 0, stream>>>(chunkP, chunkH, hInit);
  phase3_kernel<<<NBG * NCH, 256, smBytes, stream>>>(x_r, x_i, Aphase, dt_w, dt_b,
                                                     WBhi, WBlo, WChi, WClo, nlA, hInit, yout);
}
